// DimensionReductionAttention_85521388798269
// MI455X (gfx1250) — matrix operand order repaired and checked by reading, not compiled, not run on hardware
//
#include <hip/hip_runtime.h>
#include <hip/hip_bf16.h>

typedef __attribute__((ext_vector_type(16))) _Float16 v16h;
typedef __attribute__((ext_vector_type(8)))  _Float16 v8h;
typedef __attribute__((ext_vector_type(8)))  float    v8f;
typedef __attribute__((ext_vector_type(4)))  int      v4i;

#define DM     128
#define LSEQ   2048
#define TL     128
#define NTILE  (LSEQ / TL)
#define NHEAD  4
#define DH     32
#define NR     8
#define NBATCH 256  // B*C

#if defined(__has_builtin)
#if __has_builtin(__builtin_amdgcn_global_load_async_to_lds_b128)
#define USE_ASYNC_LDS 1
#endif
#endif

__device__ __forceinline__ int lane_id() { return threadIdx.x & 31; }

// 16B global -> LDS copy; async (ASYNCcnt) when the gfx1250 builtin exists.
__device__ __forceinline__ void copy16_g2l(void* lds_dst, const void* g_src) {
#ifdef USE_ASYNC_LDS
  __builtin_amdgcn_global_load_async_to_lds_b128((v4i*)g_src, (v4i*)lds_dst, 0, 0);
#else
  *(uint4*)lds_dst = *(const uint4*)g_src;
#endif
}

__device__ __forceinline__ void g2l_fence() {
#ifdef USE_ASYNC_LDS
  asm volatile("s_wait_asynccnt 0" ::: "memory");
#endif
}

__device__ __forceinline__ v8f wmma16(v16h a, v16h b, v8f c) {
  // D = A(16x32 f16) * B(32x16 f16) + C(16x16 f32)
  return __builtin_amdgcn_wmma_f32_16x16x32_f16(false, a, false, b, (short)0, c,
                                                false, false);
}

// ---- fragment helpers: every lane reads/writes CONTIGUOUS runs ----

// A tile (16 rows, 32 K) from row-major [M][K] source: 2x 16B runs per lane.
__device__ __forceinline__ v16h load_a_row(const _Float16* s, int row0, int col0, int ld) {
  const int l = lane_id();
  const int m = l & 15;
  const int hb = l >> 4;
  const _Float16* base = s + (size_t)(row0 + m) * ld + col0 + 8 * hb;
  const v8h lo = *(const v8h*)(base);
  const v8h hi = *(const v8h*)(base + 16);
  v16h a;
#pragma unroll
  for (int e = 0; e < 8; ++e) { a[e] = lo[e]; a[8 + e] = hi[e]; }
  return a;
}

// B tile 32(K) x 16(N) from [N][K]-major source: 2x 16B runs per lane.
__device__ __forceinline__ v16h load_b_nk(const _Float16* s, int k0, int n0, int ld) {
  const int l = lane_id();
  const int n = l & 15;
  const int hb = l >> 4;
  const _Float16* base = s + (size_t)(n0 + n) * ld + k0 + 8 * hb;
  const v8h lo = *(const v8h*)(base);
  const v8h hi = *(const v8h*)(base + 16);
  v16h b;
#pragma unroll
  for (int e = 0; e < 8; ++e) { b[e] = lo[e]; b[8 + e] = hi[e]; }
  return b;
}

// store D(16x16) transposed into [N][M]-major dest: 16B contiguous per lane.
__device__ __forceinline__ void store_d_f16_t(_Float16* s, v8f d, int m0, int n0, int ld) {
  const int l = lane_id();
  const int n = l & 15;
  const int hb = l >> 4;
  v8h v;
#pragma unroll
  for (int j = 0; j < 8; ++j) v[j] = (_Float16)d[j];
  *(v8h*)(s + (size_t)(n0 + n) * ld + m0 + 8 * hb) = v;
}

// strided row-major stores (only for small / epilogue tiles)
__device__ __forceinline__ void store_d_f16_row(_Float16* s, v8f d, int row0, int col0, int ld) {
  const int l = lane_id();
  const int n = l & 15;
  const int hb = l >> 4;
#pragma unroll
  for (int j = 0; j < 8; ++j)
    s[(size_t)(row0 + j + 8 * hb) * ld + col0 + n] = (_Float16)d[j];
}

__device__ __forceinline__ void store_d_f32_row(float* s, v8f d, int row0, int col0, int ld) {
  const int l = lane_id();
  const int n = l & 15;
  const int hb = l >> 4;
#pragma unroll
  for (int j = 0; j < 8; ++j)
    s[(size_t)(row0 + j + 8 * hb) * ld + col0 + n] = d[j];
}

// Y[l][d] = X(l,k) @ W(k,d) + bias(d), computed as Wt-strip x X, t-store.
__device__ __forceinline__ void gemm_wtx(const _Float16* Wt, const _Float16* X,
                                         const float* bias, _Float16* Y, int wave) {
  const int d0 = wave * 16;
  const int hb = (threadIdx.x >> 4) & 1;
  v16h a[4];
#pragma unroll
  for (int kt = 0; kt < 4; ++kt) a[kt] = load_a_row(Wt, d0, kt * 32, DM);
  float bvals[8];
#pragma unroll
  for (int j = 0; j < 8; ++j) bvals[j] = bias[d0 + 8 * hb + j];
#pragma unroll
  for (int nt = 0; nt < 8; ++nt) {
    v8f acc = {};
#pragma unroll
    for (int kt = 0; kt < 4; ++kt)
      acc = wmma16(a[kt], load_b_nk(X, kt * 32, nt * 16, DM), acc);
#pragma unroll
    for (int j = 0; j < 8; ++j) acc[j] += bvals[j];
    store_d_f16_t(Y, acc, d0, nt * 16, DM);  // Y[l][d]
  }
}

// Yt[d][l] = (X @ W + bias)^T via A = X row-major strip over l, B = Wt, t-store.
__device__ __forceinline__ void gemm_xw_t(const _Float16* X, const _Float16* Wt,
                                          const float* bias, _Float16* Yt, int wave) {
  const int l0 = wave * 16;
  const int nl = threadIdx.x & 15;
  v16h a[4];
#pragma unroll
  for (int kt = 0; kt < 4; ++kt) a[kt] = load_a_row(X, l0, kt * 32, DM);
#pragma unroll
  for (int nt = 0; nt < 8; ++nt) {
    v8f acc = {};
#pragma unroll
    for (int kt = 0; kt < 4; ++kt)
      acc = wmma16(a[kt], load_b_nk(Wt, kt * 32, nt * 16, DM), acc);
    const float bv = bias[nt * 16 + nl];
#pragma unroll
    for (int j = 0; j < 8; ++j) acc[j] += bv;
    store_d_f16_t(Yt, acc, l0, nt * 16, DM);  // Yt[d][l]
  }
}

// ------------------------- prep kernels -------------------------

// store W transposed as f16: dst[n*128+k] = src[k*128+n]
__global__ __launch_bounds__(256) void dra_cvt_t(const float* __restrict__ src,
                                                 _Float16* __restrict__ dst) {
  const int i = blockIdx.x * 256 + threadIdx.x;
  const int n = i >> 7, k = i & 127;
  dst[i] = (_Float16)src[k * DM + n];
}

// q1 = router @ Wq1 + bq1, per-head padded 16x32 A-tiles (rows 8..15 zero)
__global__ __launch_bounds__(256) void dra_q1(const float* __restrict__ router,
                                              const float* __restrict__ Wq1,
                                              const float* __restrict__ bq1,
                                              _Float16* __restrict__ q1h) {
  const int t = threadIdx.x;
  for (int i = t; i < NHEAD * 16 * DH; i += 256) {
    const int h = i >> 9, rem = i & 511, r = rem >> 5, d = rem & 31;
    float v = 0.f;
    if (r < NR) {
      const int c = h * DH + d;
      v = bq1[c];
      for (int k = 0; k < DM; ++k) v += router[r * DM + k] * Wq1[k * DM + c];
    }
    q1h[i] = (_Float16)v;
  }
}

// ------------------------- stage 1 -------------------------
__global__ __launch_bounds__(256) void dra_stage1(
    const float* __restrict__ Z,
    const float* __restrict__ bk1, const float* __restrict__ bv1,
    const float* __restrict__ Wo1, const float* __restrict__ bo1,
    const float* __restrict__ Wk2, const float* __restrict__ bk2,
    const float* __restrict__ Wv2, const float* __restrict__ bv2,
    const _Float16* __restrict__ Wt8h, const _Float16* __restrict__ q1h,
    _Float16* __restrict__ k2t, _Float16* __restrict__ v2p) {
  extern __shared__ char smem_raw[];
  _Float16* sWkT = (_Float16*)smem_raw;  // Wk1^T [d][k]
  _Float16* sWvT = sWkT + 16384;         // Wv1^T [d][k]
  _Float16* sX   = sWvT + 16384;         // x tile [l][k]
  _Float16* sK   = sX + 16384;           // K tile [l][d]
  _Float16* sVt  = sK + 16384;           // V tile transposed [d][l]
  _Float16* sP   = sVt + 16384;          // 8 waves * 16x32, P [q][l]
  float* sO   = (float*)(sP + 4096);     // 8 waves * 16x32 f32, O [q][d]
  float* sSum = sO + 4096;               // 8 waves * 16
  float* sRB  = sSum + 128;              // 8x128 attn out
  float* sRB2 = sRB + 1024;              // 8x128 router_buffer

  const int n = blockIdx.x;
  const int t = threadIdx.x;
  const int wave = t >> 5;
  const int h = wave >> 1;  // two waves per head

  {  // stage transposed Wk1 (slot 1), Wv1 (slot 2) -- async when available
    const uint4* gk = (const uint4*)(Wt8h + 1 * 16384);
    const uint4* gv = (const uint4*)(Wt8h + 2 * 16384);
    uint4* dk = (uint4*)sWkT;
    uint4* dv = (uint4*)sWvT;
    for (int i = t; i < 2048; i += 256) {
      copy16_g2l(dk + i, gk + i);
      copy16_g2l(dv + i, gv + i);
    }
  }

  const v16h qa = load_a_row(q1h + h * 512, 0, 0, DH);  // per-head query A-tile

  v8f oacc0 = {}, oacc1 = {};
  float srow[8];
#pragma unroll
  for (int j = 0; j < 8; ++j) srow[j] = 0.f;

  const float scale = 0.1767766952966369f;  // 1/sqrt(32)
  const float* xg = Z + (size_t)n * LSEQ * DM;
  const int nl = t & 15;
  const int hb = (t >> 4) & 1;

  g2l_fence();
  __syncthreads();

  for (int lt = 0; lt < NTILE; ++lt) {
    const float* xt = xg + (size_t)lt * TL * DM;
    {  // load + convert x tile fp32 -> f16
      const float4* xs = (const float4*)xt;
      for (int i = t; i < 4096; i += 256) {
        const float4 v = xs[i];
        _Float16* d = sX + 4 * i;
        d[0] = (_Float16)v.x; d[1] = (_Float16)v.y;
        d[2] = (_Float16)v.z; d[3] = (_Float16)v.w;
      }
    }
    __syncthreads();

    gemm_wtx(sWkT, sX, bk1, sK, wave);   // K tile [l][d]
    gemm_xw_t(sX, sWvT, bv1, sVt, wave); // V tile [d][l]
    __syncthreads();

    _Float16* pw = sP + wave * 512;  // this wave's 16x32 P tile [q][l]
#pragma unroll
    for (int pp = 0; pp < 2; ++pp) {
      const int c0 = (wave & 1) * 64 + pp * 32;  // 32 L-rows per chunk pair
      v8f s0 = {}, s1 = {};
      // scores: B(k=d, n=l) = sK[(c0+n)][h*32+k]  (contiguous per lane)
      s0 = wmma16(qa, load_b_nk(sK, h * DH, c0, DM), s0);
      s1 = wmma16(qa, load_b_nk(sK, h * DH, c0 + 16, DM), s1);
#pragma unroll
      for (int j = 0; j < 8; ++j) {
        const float e0 = __expf(s0[j] * scale);
        const float e1 = __expf(s1[j] * scale);
        srow[j] += e0 + e1;
        pw[(j + 8 * hb) * 32 + nl]      = (_Float16)e0;
        pw[(j + 8 * hb) * 32 + 16 + nl] = (_Float16)e1;
      }
      const v16h pa = load_a_row(pw, 0, 0, 32);
      // O += P @ V : B(k=l, n=d) = sVt[(h*32+n)][c0+k]  (contiguous per lane)
      oacc0 = wmma16(pa, load_b_nk(sVt, c0, h * DH, DM), oacc0);
      oacc1 = wmma16(pa, load_b_nk(sVt, c0, h * DH + 16, DM), oacc1);
    }

    if (lt + 1 < NTILE) {  // prefetch next x tile (global_prefetch_b8)
      const float* nx = xt + TL * DM;
      for (int i = t * 16; i < TL * DM; i += 4096) __builtin_prefetch(nx + i, 0, 1);
    }
    __syncthreads();
  }

  // row-sum reduction across the 16 lanes holding each row
#pragma unroll
  for (int j = 0; j < 8; ++j) {
    float s = srow[j];
    s += __shfl_xor(s, 1, 32);
    s += __shfl_xor(s, 2, 32);
    s += __shfl_xor(s, 4, 32);
    s += __shfl_xor(s, 8, 32);
    srow[j] = s;
  }
  {
    float* ow = sO + wave * 512;
    store_d_f32_row(ow, oacc0, 0, 0, 32);
    store_d_f32_row(ow, oacc1, 0, 16, 32);
    if ((t & 15) == 0) {
#pragma unroll
      for (int j = 0; j < 8; ++j) sSum[wave * 16 + j + 8 * hb] = srow[j];
    }
  }
  __syncthreads();

  // combine the two waves per head, normalize: attn (8 x 128)
  for (int i = t; i < NR * DM; i += 256) {
    const int r = i >> 7, c = i & 127, hh = c >> 5, d = c & 31;
    const float num = sO[(2 * hh) * 512 + r * 32 + d] + sO[(2 * hh + 1) * 512 + r * 32 + d];
    const float den = sSum[(2 * hh) * 16 + r] + sSum[(2 * hh + 1) * 16 + r];
    sRB[i] = num / den;
  }
  __syncthreads();

  // router_buffer = attn @ Wo1 + bo1  (tiny 8x128 GEMM, fp32)
  for (int i = t; i < NR * DM; i += 256) {
    const int r = i >> 7, c = i & 127;
    float acc = bo1[c];
    for (int k = 0; k < DM; ++k) acc += sRB[r * DM + k] * Wo1[k * DM + c];
    sRB2[i] = acc;
  }
  __syncthreads();

  // k2 -> [h][r(16)][d(32)] (rows r>=8 zero): contiguous B for stage2 scores
  // v2 -> [h][d(32)][r(32 pad)]            : contiguous B for stage2 A*V
  _Float16* k2n = k2t + (size_t)n * 2048;
  _Float16* v2n = v2p + (size_t)n * 4096;
  for (int i = t; i < 2048; i += 256) {
    const int hh = i >> 9, rem = i & 511, r = rem >> 5, d = rem & 31;
    float val = 0.f;
    if (r < NR) {
      const int c = hh * DH + d;
      val = bk2[c];
      for (int k = 0; k < DM; ++k) val += sRB2[r * DM + k] * Wk2[k * DM + c];
    }
    k2n[i] = (_Float16)val;
  }
  for (int i = t; i < 4096; i += 256) {
    const int hh = i >> 10, rem = i & 1023, d = rem >> 5, r = rem & 31;
    float val = 0.f;
    if (r < NR) {
      const int c = hh * DH + d;
      val = bv2[c];
      for (int k = 0; k < DM; ++k) val += sRB2[r * DM + k] * Wv2[k * DM + c];
    }
    v2n[i] = (_Float16)val;
  }
}

// ------------------------- stage 2 -------------------------
__global__ __launch_bounds__(256) void dra_stage2(
    const float* __restrict__ Z,
    const float* __restrict__ bq2, const float* __restrict__ bo2,
    const float* __restrict__ gamma, const float* __restrict__ beta,
    const _Float16* __restrict__ Wt8h,
    const _Float16* __restrict__ k2t, const _Float16* __restrict__ v2p,
    float* __restrict__ out) {
  extern __shared__ char smem_raw[];
  _Float16* sWqT = (_Float16*)smem_raw;  // Wq2^T [d][k]  \ reused later as
  _Float16* sX   = sWqT + 16384;         // x tile [l][k] / 128x128 f32 (sY)
  _Float16* sQ   = sX + 16384;           // q2 [l][d], then attn-out in place
  _Float16* sWoT = sQ + 16384;           // Wo2^T [d][k]
  _Float16* sK2  = sWoT + 16384;         // [h][r(16)][d(32)]
  _Float16* sV2  = sK2 + 2048;           // [h][d(32)][r(32)]
  _Float16* sP   = sV2 + 4096;           // 8 waves * 16x32, P [l][r]
  float* sMu   = (float*)(sP + 4096);    // 128
  float* sRstd = sMu + 128;              // 128
  float* sY = (float*)smem_raw;          // 16384 f32, aliases [sWqT|sX]

  const int blk = blockIdx.x;
  const int n = blk >> 4;
  const int lt = blk & 15;
  const int t = threadIdx.x;
  const int wave = t >> 5;
  const int r0 = wave * 16;

  {  // stage Wq2^T (slot 4), Wo2^T (slot 7), k2/v2 -- async when available
    const uint4* gq = (const uint4*)(Wt8h + 4 * 16384);
    const uint4* go = (const uint4*)(Wt8h + 7 * 16384);
    uint4* dq = (uint4*)sWqT;
    uint4* dw = (uint4*)sWoT;
    for (int i = t; i < 2048; i += 256) {
      copy16_g2l(dq + i, gq + i);
      copy16_g2l(dw + i, go + i);
    }
    const uint4* gk = (const uint4*)(k2t + (size_t)n * 2048);
    const uint4* gv = (const uint4*)(v2p + (size_t)n * 4096);
    uint4* dk = (uint4*)sK2;
    uint4* dv = (uint4*)sV2;
    for (int i = t; i < 256; i += 256) copy16_g2l(dk + i, gk + i);
    for (int i = t; i < 512; i += 256) copy16_g2l(dv + i, gv + i);
  }
  for (int i = t; i < 4096; i += 256) sP[i] = (_Float16)0.f;  // pad cols stay 0

  const float* xt = Z + ((size_t)n * LSEQ + (size_t)lt * TL) * DM;
  {  // overlaps with async staging above
    const float4* xs = (const float4*)xt;
    for (int i = t; i < 4096; i += 256) {
      const float4 v = xs[i];
      _Float16* d = sX + 4 * i;
      d[0] = (_Float16)v.x; d[1] = (_Float16)v.y;
      d[2] = (_Float16)v.z; d[3] = (_Float16)v.w;
    }
  }
  g2l_fence();
  __syncthreads();

  gemm_wtx(sWqT, sX, bq2, sQ, wave);  // q2 [l][d] (wave strip over d)
  __syncthreads();                    // q2 strips cross wave boundaries below

  const float scale = 0.1767766952966369f;
  _Float16* pw = sP + wave * 512;
  const int nl = t & 15;
  const int hb = (t >> 4) & 1;

#pragma unroll
  for (int h = 0; h < NHEAD; ++h) {
    const v16h qfr = load_a_row(sQ, r0, h * DH, DM);  // A(m=l, k=d)
    v8f sc = {};
    // B(k=d, n=r) = sK2[h][n][k]  (contiguous per lane)
    sc = wmma16(qfr, load_b_nk(sK2 + h * 512, 0, 0, DH), sc);
    // softmax over the 8 routers (cols 8..15 padded zero)
#pragma unroll
    for (int j = 0; j < 8; ++j) {
      const float ev = (nl < NR) ? __expf(sc[j] * scale) : 0.f;
      float s = ev;
      s += __shfl_xor(s, 1, 32);
      s += __shfl_xor(s, 2, 32);
      s += __shfl_xor(s, 4, 32);
      s += __shfl_xor(s, 8, 32);
      pw[(j + 8 * hb) * 32 + nl] = (_Float16)(ev / s);  // P[l][r]
    }
    const v16h pa = load_a_row(pw, 0, 0, 32);
    v8f o0 = {}, o1 = {};
    // B(k=r, n=d) = sV2[h][n][k]  (contiguous per lane)
    o0 = wmma16(pa, load_b_nk(sV2 + h * 1024, 0, 0, 32), o0);
    o1 = wmma16(pa, load_b_nk(sV2 + h * 1024, 0, 16, 32), o1);
    // overwrite q2 columns of this head in place (wave-local rows)
    store_d_f16_row(sQ, o0, r0, h * DH, DM);
    store_d_f16_row(sQ, o1, r0, h * DH + 16, DM);
  }
  __syncthreads();

  {  // out2 = attn_out @ Wo2, f32 into sY (sWqT/sX are dead now)
    v16h a[4];
#pragma unroll
    for (int kt = 0; kt < 4; ++kt) a[kt] = load_a_row(sQ, r0, kt * 32, DM);
#pragma unroll
    for (int nt = 0; nt < 8; ++nt) {
      v8f acc = {};
#pragma unroll
      for (int kt = 0; kt < 4; ++kt)
        acc = wmma16(a[kt], load_b_nk(sWoT, kt * 32, nt * 16, DM), acc);
      store_d_f32_row(sY, acc, r0, nt * 16, DM);
    }
  }
  __syncthreads();

  // residual + output bias
  for (int i = t; i < TL * DM; i += 256) {
    const int c = i & 127;
    sY[i] += xt[i] + bo2[c];
  }
  __syncthreads();

  // per-row layernorm stats
  if (t < TL) {
    float mu = 0.f;
    for (int c = 0; c < DM; ++c) mu += sY[t * DM + c];
    mu *= (1.0f / DM);
    float var = 0.f;
    for (int c = 0; c < DM; ++c) { const float d = sY[t * DM + c] - mu; var += d * d; }
    var *= (1.0f / DM);
    sMu[t] = mu;
    sRstd[t] = rsqrtf(var + 1e-5f);
  }
  __syncthreads();

  float* og = out + ((size_t)n * LSEQ + (size_t)lt * TL) * DM;
  for (int i = t; i < TL * DM; i += 256) {
    const int r = i >> 7, c = i & 127;
    og[i] = (sY[i] - sMu[r]) * sRstd[r] * gamma[c] + beta[c];
  }
}

// ------------------------- launcher -------------------------

extern "C" void kernel_launch(void* const* d_in, const int* in_sizes, int n_in,
                              void* d_out, int out_size, void* d_ws, size_t ws_size,
                              hipStream_t stream) {
  (void)in_sizes; (void)n_in; (void)out_size; (void)ws_size;
  const float* Z      = (const float*)d_in[0];
  const float* router = (const float*)d_in[1];
  const float* Wq1 = (const float*)d_in[2];  const float* bq1 = (const float*)d_in[3];
  const float* Wk1 = (const float*)d_in[4];  const float* bk1 = (const float*)d_in[5];
  const float* Wv1 = (const float*)d_in[6];  const float* bv1 = (const float*)d_in[7];
  const float* Wo1 = (const float*)d_in[8];  const float* bo1 = (const float*)d_in[9];
  const float* Wq2 = (const float*)d_in[10]; const float* bq2 = (const float*)d_in[11];
  const float* Wk2 = (const float*)d_in[12]; const float* bk2 = (const float*)d_in[13];
  const float* Wv2 = (const float*)d_in[14]; const float* bv2 = (const float*)d_in[15];
  const float* Wo2 = (const float*)d_in[16]; const float* bo2 = (const float*)d_in[17];
  const float* gamma = (const float*)d_in[18];
  const float* beta  = (const float*)d_in[19];
  float* out = (float*)d_out;

  // workspace (f16): 8 transposed weight mats | q1 tiles | k2 | v2 (~3.3 MB)
  _Float16* Wt8h = (_Float16*)d_ws;
  _Float16* q1h  = Wt8h + 8 * 16384;
  _Float16* k2t  = q1h + 2048;
  _Float16* v2p  = k2t + (size_t)NBATCH * 2048;

  const float* Ws[8] = {Wq1, Wk1, Wv1, Wo1, Wq2, Wk2, Wv2, Wo2};
  for (int m = 0; m < 8; ++m)
    dra_cvt_t<<<64, 256, 0, stream>>>(Ws[m], Wt8h + m * 16384);
  dra_q1<<<1, 256, 0, stream>>>(router, Wq1, bq1, q1h);

  const size_t smem1 = (size_t)(5 * 16384 + 4096) * sizeof(_Float16) +
                       (size_t)(4096 + 128 + 1024 + 1024) * sizeof(float);
  dra_stage1<<<NBATCH, 256, smem1, stream>>>(Z, bk1, bv1, Wo1, bo1,
                                             Wk2, bk2, Wv2, bv2,
                                             Wt8h, q1h, k2t, v2p);

  const size_t smem2 = (size_t)(4 * 16384 + 2048 + 4096 + 4096) * sizeof(_Float16) +
                       (size_t)256 * sizeof(float);
  dra_stage2<<<NBATCH * NTILE, 256, smem2, stream>>>(Z, bq2, bo2, gamma, beta,
                                                     Wt8h, k2t, v2p, out);
}
